// MultiScaleAttention_8813272891950
// MI455X (gfx1250) — compile-verified, matrix operand order checked by reading, not run on hardware
//
#include <hip/hip_runtime.h>
#include <hip/hip_bf16.h>
#include <math.h>
#include <stdint.h>

typedef __attribute__((ext_vector_type(16))) _Float16 v16h;
typedef __attribute__((ext_vector_type(8)))  _Float16 v8h;
typedef __attribute__((ext_vector_type(8)))  float    v8f;
typedef __attribute__((ext_vector_type(4)))  unsigned int v4u;
typedef __attribute__((ext_vector_type(8)))  int      v8i;
typedef __attribute__((ext_vector_type(4)))  int      v4i;

#define DIM    256
#define NHEADS 8
#define BATCH  8
#define HIMG   64
#define NTOK   (HIMG * HIMG)
#define MGRID  32

// ---------------------------------------------------------------------------
// WMMA helpers (CDNA5 gfx1250, wave32).
// A operand (16x32 f16): lane m=lane&15 is row M, h=lane>>4; element e maps to
// K = (e/8)*16 + h*8 + (e%8)  => two contiguous 8-half (16B) loads per lane.
// B operand is the same pattern applied to the *transposed* [N][K] storage.
// C/D (16x16 f32): element r -> row = r + 8*(lane>>4), col = lane&15.
// ---------------------------------------------------------------------------
__device__ __forceinline__ v8f wmma_f16(v16h a, v16h b, v8f c) {
  return __builtin_amdgcn_wmma_f32_16x16x32_f16(
      /*neg_a=*/false, a, /*neg_b=*/false, b,
      /*c_mod=*/(short)0, c, /*reuse_a=*/false, /*reuse_b=*/false);
}

__device__ __forceinline__ v16h load_frag(const _Float16* base, int ldh, int lane) {
  int m = lane & 15, h = lane >> 4;
  const _Float16* row = base + (size_t)m * (size_t)ldh;
  v8h lo = *(const v8h*)(row + h * 8);
  v8h hi = *(const v8h*)(row + 16 + h * 8);
  v16h r;
#pragma unroll
  for (int i = 0; i < 8; ++i) { r[i] = lo[i]; r[8 + i] = hi[i]; }
  return r;
}

// Same, but only K=0..15 are real (K=16..31 forced to zero, no OOB load).
__device__ __forceinline__ v16h load_frag_lo(const _Float16* base, int ldh, int lane) {
  int m = lane & 15, h = lane >> 4;
  const _Float16* row = base + (size_t)m * (size_t)ldh;
  v8h lo = *(const v8h*)(row + h * 8);
  v16h r;
#pragma unroll
  for (int i = 0; i < 8; ++i) { r[i] = lo[i]; r[8 + i] = (_Float16)0.f; }
  return r;
}

// ---------------------------------------------------------------------------
// Tensor Data Mover: DMA one 2-D f16 tile (tile_rows x tile_k elements, row
// stride = row_stride elements) from global memory into LDS at lds_off.
// D# built per CDNA5 ISA 08_async_tensor: group0 = {count=1, lds_addr,
// global_addr[56:0], type=2}; group1 = {data_size=2B, tensor_dim0/1 = tile
// dims (tile always fully in-bounds), tile_dim0/1, tensor_dim0_stride}.
// Tracked on TENSORcnt.
// ---------------------------------------------------------------------------
__device__ __forceinline__ void tdm_load_2d(unsigned lds_off, const _Float16* gptr,
                                            unsigned tile_k, unsigned tile_rows,
                                            unsigned row_stride) {
  unsigned long long ga = (unsigned long long)(size_t)gptr;
  v4u g0;
  g0[0] = 1u;                                   // count=1 (valid), user mode
  g0[1] = lds_off;                              // LDS byte address
  g0[2] = (unsigned)(ga & 0xffffffffu);         // global_addr[31:0]
  g0[3] = (unsigned)((ga >> 32) & 0x01ffffffu)  // global_addr[56:32]
          | (2u << 30);                         // type = 2 ("image")
  v8i g1;
  g1[0] = (int)(1u << 16);                                      // data_size=1 -> 2 bytes
  g1[1] = (int)((tile_k & 0xffffu) << 16);                      // tensor_dim0 lo16
  g1[2] = (int)((tile_k >> 16) | ((tile_rows & 0xffffu) << 16)); // dim0 hi | dim1 lo
  g1[3] = (int)((tile_rows >> 16) | ((tile_k & 0xffffu) << 16)); // dim1 hi | tile_dim0
  g1[4] = (int)(tile_rows & 0xffffu);                           // tile_dim1 | tile_dim2=0
  g1[5] = (int)row_stride;                                      // tensor_dim0_stride lo32
  g1[6] = 0;                                                    // stride0 hi | stride1 lo
  g1[7] = 0;
  v4i z4 = {0, 0, 0, 0};
#if defined(__clang_major__) && (__clang_major__ >= 23)
  v8i z8 = {0, 0, 0, 0, 0, 0, 0, 0};
  __builtin_amdgcn_tensor_load_to_lds(g0, g1, z4, z4, z8, 0);
#else
  __builtin_amdgcn_tensor_load_to_lds(g0, g1, z4, z4, 0);
#endif
}

// ---------------------------------------------------------------------------
// f32 -> f16 convert (row-major copy).
// ---------------------------------------------------------------------------
__global__ void cvt_f16_kernel(const float* __restrict__ src,
                               _Float16* __restrict__ dst, int tot) {
  int i = blockIdx.x * blockDim.x + threadIdx.x;
  if (i < tot) dst[i] = (_Float16)src[i];
}

// Wt[n][k] = (f16) W[k][n]  (W is [K][N] f32 row-major).
__global__ void transpose_f16_kernel(const float* __restrict__ W,
                                     _Float16* __restrict__ Wt,
                                     int K, int N, int tot) {
  int i = blockIdx.x * blockDim.x + threadIdx.x;
  if (i >= tot) return;
  int k = i % K, n = i / K;
  Wt[(size_t)n * K + k] = (_Float16)W[(size_t)k * N + n];
}

// ---------------------------------------------------------------------------
// Tiled WMMA GEMM with TDM staging: C[M,N] = A[M,K] * B[K,N] (+bias).
// A is f16 row-major [M][K]; Bt is f16 transposed [N][K]; C fp32.
// Block = 256 threads (8 waves), tile 128x64, K-step 32. Wave 0 DMAs both
// tiles into LDS via tensor_load_to_lds and waits on TENSORcnt; the barrier
// publishes them. Wave w owns rows 16w..16w+15 and all 64 cols (4 WMMAs/step).
// ---------------------------------------------------------------------------
__global__ __launch_bounds__(256) void wmma_gemm_f16_kernel(
    const _Float16* __restrict__ A, const _Float16* __restrict__ Bt,
    const float* __restrict__ bias, float* __restrict__ C,
    int M, int N, int K) {
  __shared__ __align__(16) _Float16 As[128][32];
  __shared__ __align__(16) _Float16 Bs[64][32];   // Bs[n][k]
  (void)M;

  int tid = threadIdx.x;
  int lane = tid & 31, wave = tid >> 5;
  int row0 = blockIdx.y * 128;
  int col0 = blockIdx.x * 64;

  v8f acc[4];
#pragma unroll
  for (int t = 0; t < 4; ++t)
#pragma unroll
    for (int r = 0; r < 8; ++r) acc[t][r] = 0.f;

  // LDS byte offsets (low 32 bits of the flat shared address).
  unsigned lds_a = (unsigned)(size_t)&As[0][0];
  unsigned lds_b = (unsigned)(size_t)&Bs[0][0];

  for (int k0 = 0; k0 < K; k0 += 32) {
    if (wave == 0) {
      tdm_load_2d(lds_a, A + (size_t)row0 * (size_t)K + k0, 32u, 128u, (unsigned)K);
      tdm_load_2d(lds_b, Bt + (size_t)col0 * (size_t)K + k0, 32u, 64u, (unsigned)K);
      __builtin_amdgcn_s_wait_tensorcnt(0);
    }
    __syncthreads();

    v16h af = load_frag(&As[wave * 16][0], 32, lane);
#pragma unroll
    for (int t = 0; t < 4; ++t) {
      v16h bf = load_frag(&Bs[t * 16][0], 32, lane);
      acc[t] = wmma_f16(af, bf, acc[t]);
    }
    __syncthreads();  // protect LDS from next TDM overwrite
  }

  int n = lane & 15, hh = lane >> 4;
#pragma unroll
  for (int t = 0; t < 4; ++t) {
#pragma unroll
    for (int r = 0; r < 8; ++r) {
      int row = row0 + wave * 16 + r + 8 * hh;
      int col = col0 + t * 16 + n;
      float v = acc[t][r];
      if (bias) v += bias[col];
      C[(size_t)row * (size_t)N + col] = v;
    }
  }
}

// ---------------------------------------------------------------------------
// Pack per-head Q/K (f16, patch-order [B][NP][P2][32]) and V^T ([B][NP][32][P2]).
// ---------------------------------------------------------------------------
__global__ void pack_head_kernel(
    const float* __restrict__ Q, const float* __restrict__ KV, int ldkv,
    int qcol, int kcol, int vcol,
    _Float16* __restrict__ qh, _Float16* __restrict__ kh, _Float16* __restrict__ vth,
    int Hs, int p, int tot) {
  int idx = blockIdx.x * blockDim.x + threadIdx.x;
  if (idx >= tot) return;
  int P2 = p * p, npx = Hs / p, NP = npx * npx;
  int ch = idx & 31;
  int t = idx >> 5;
  int within = t % P2; t /= P2;
  int patch = t % NP;
  int b = t / NP;
  int py = patch / npx, px = patch % npx;
  int wy = within / p, wx = within % p;
  int nsp = (py * p + wy) * Hs + (px * p + wx);
  size_t srow = (size_t)b * (size_t)(Hs * Hs) + (size_t)nsp;
  float qv = Q[srow * 256 + qcol + ch];
  float kv = KV[srow * (size_t)ldkv + kcol + ch];
  float vv = KV[srow * (size_t)ldkv + vcol + ch];
  size_t o = ((size_t)(b * NP + patch) * P2 + within) * 32 + ch;
  qh[o] = (_Float16)qv;
  kh[o] = (_Float16)kv;
  vth[((size_t)(b * NP + patch) * 32 + ch) * (size_t)P2 + within] = (_Float16)vv;
}

// ---------------------------------------------------------------------------
// Flash-style local attention (one 16-row query tile per wave).
// ---------------------------------------------------------------------------
__global__ __launch_bounds__(256) void attn_wmma_kernel(
    const _Float16* __restrict__ qh, const _Float16* __restrict__ kh,
    const _Float16* __restrict__ vth, float* __restrict__ out,
    int head_col, int P2, int NP, int tokN, float scale, int total_tiles) {
  __shared__ __align__(16) _Float16 pl[8][16][32];

  int lane = threadIdx.x & 31, wave = threadIdx.x >> 5;
  int tile = blockIdx.x * 8 + wave;
  if (tile >= total_tiles) return;  // no __syncthreads in this kernel

  int ntq = P2 >> 4;
  int b = tile / (NP * ntq);
  int rem = tile % (NP * ntq);
  int patch = rem / ntq;
  int qt = rem % ntq;

  const _Float16* qbase  = qh  + ((size_t)(b * NP + patch) * P2 + (size_t)qt * 16) * 32;
  const _Float16* kpatch = kh  + (size_t)(b * NP + patch) * (size_t)P2 * 32;
  const _Float16* vpatch = vth + (size_t)(b * NP + patch) * 32 * (size_t)P2;

  v16h aq = load_frag(qbase, 32, lane);

  v8f o0, o1;
  float mrow[8], lrow[8];
#pragma unroll
  for (int r = 0; r < 8; ++r) { o0[r] = 0.f; o1[r] = 0.f; mrow[r] = -1e30f; lrow[r] = 0.f; }

  int col = lane & 15, hh = lane >> 4;

  for (int key0 = 0; key0 < P2; key0 += 32) {
    bool full = (key0 + 16) < P2;  // P2==16 head uses only the low half-chunk

    v8f zero;
#pragma unroll
    for (int r = 0; r < 8; ++r) zero[r] = 0.f;

    v16h bk0 = load_frag(kpatch + (size_t)key0 * 32, 32, lane);
    v8f s0 = wmma_f16(aq, bk0, zero);
    v8f s1;
#pragma unroll
    for (int r = 0; r < 8; ++r) s1[r] = -1e30f;
    if (full) {
      v16h bk1 = load_frag(kpatch + (size_t)(key0 + 16) * 32, 32, lane);
      s1 = wmma_f16(aq, bk1, zero);
    }

    float p0[8], p1[8];
#pragma unroll
    for (int r = 0; r < 8; ++r) {
      float a = s0[r] * scale;
      float c = full ? (s1[r] * scale) : -1e30f;
      float mx = fmaxf(a, c);
#pragma unroll
      for (int m = 1; m < 16; m <<= 1) mx = fmaxf(mx, __shfl_xor(mx, m, 32));
      float mnew = fmaxf(mrow[r], mx);
      float alpha = __expf(mrow[r] - mnew);
      mrow[r] = mnew;
      p0[r] = __expf(a - mnew);
      p1[r] = __expf(c - mnew);
      float rs = p0[r] + p1[r];
#pragma unroll
      for (int m = 1; m < 16; m <<= 1) rs += __shfl_xor(rs, m, 32);
      lrow[r] = lrow[r] * alpha + rs;
      o0[r] *= alpha;
      o1[r] *= alpha;
    }

    // transpose P (C/D layout -> A layout) through per-wave LDS tile
#pragma unroll
    for (int r = 0; r < 8; ++r) {
      int row = r + 8 * hh;
      pl[wave][row][col]      = (_Float16)p0[r];
      pl[wave][row][16 + col] = (_Float16)p1[r];
    }
    __builtin_amdgcn_wave_barrier();  // same-wave DS ops are in-order
    v16h ap = load_frag(&pl[wave][0][0], 32, lane);

    v16h bv0, bv1;
    if (full) {
      bv0 = load_frag(vpatch + key0, P2, lane);
      bv1 = load_frag(vpatch + (size_t)16 * P2 + key0, P2, lane);
    } else {
      bv0 = load_frag_lo(vpatch + key0, P2, lane);
      bv1 = load_frag_lo(vpatch + (size_t)16 * P2 + key0, P2, lane);
    }
    o0 = wmma_f16(ap, bv0, o0);
    o1 = wmma_f16(ap, bv1, o1);
  }

#pragma unroll
  for (int r = 0; r < 8; ++r) {
    int row = qt * 16 + r + 8 * hh;  // within patch
    size_t nidx = (size_t)b * (size_t)tokN + (size_t)patch * P2 + row;
    float inv = 1.f / lrow[r];
    out[nidx * 256 + head_col + col]      = o0[r] * inv;
    out[nidx * 256 + head_col + 16 + col] = o1[r] * inv;
  }
}

// ---------------------------------------------------------------------------
// Global-query helpers.
// ---------------------------------------------------------------------------
__global__ void qg_mean_kernel(const float* __restrict__ Q, int qcol,
                               float* __restrict__ qg, int Hs, int p, int tot) {
  int idx = blockIdx.x * blockDim.x + threadIdx.x;
  if (idx >= tot) return;
  int P2 = p * p, npx = Hs / p, NP = npx * npx;
  int ch = idx & 31;
  int t = idx >> 5;
  int within = t % P2;
  int b = t / P2;
  int wy = within / p, wx = within % p;
  float s = 0.f;
  for (int pp = 0; pp < NP; ++pp) {
    int py = pp / npx, px = pp % npx;
    int nsp = (py * p + wy) * Hs + (px * p + wx);
    s += Q[((size_t)b * (size_t)(Hs * Hs) + nsp) * 256 + qcol + ch];
  }
  qg[((size_t)b * P2 + within) * 32 + ch] = s / (float)NP;
}

__global__ void qg_resize_kernel(const float* __restrict__ qg, float* __restrict__ Qg,
                                 int qcol, int p, int tot) {
  int idx = blockIdx.x * blockDim.x + threadIdx.x;
  if (idx >= tot) return;
  int ch = idx & 31;
  int t = idx >> 5;
  int pos = t % (MGRID * MGRID);
  int b = t / (MGRID * MGRID);
  int my = pos >> 5, mx = pos & 31;
  float sc = (float)p / (float)MGRID;
  float sy = fminf(fmaxf((my + 0.5f) * sc - 0.5f, 0.f), (float)(p - 1));
  float sx = fminf(fmaxf((mx + 0.5f) * sc - 0.5f, 0.f), (float)(p - 1));
  int y0 = (int)floorf(sy); int y1 = y0 + 1 < p ? y0 + 1 : p - 1; float fy = sy - (float)y0;
  int x0 = (int)floorf(sx); int x1 = x0 + 1 < p ? x0 + 1 : p - 1; float fx = sx - (float)x0;
  size_t base = (size_t)b * (size_t)(p * p);
  float v00 = qg[(base + y0 * p + x0) * 32 + ch];
  float v01 = qg[(base + y0 * p + x1) * 32 + ch];
  float v10 = qg[(base + y1 * p + x0) * 32 + ch];
  float v11 = qg[(base + y1 * p + x1) * 32 + ch];
  float v = (1.f - fy) * ((1.f - fx) * v00 + fx * v01) + fy * ((1.f - fx) * v10 + fx * v11);
  Qg[((size_t)b * (MGRID * MGRID) + pos) * 256 + qcol + ch] = v;
}

__global__ void pool2x2_kernel(const float* __restrict__ A, float* __restrict__ P, int tot) {
  int idx = blockIdx.x * blockDim.x + threadIdx.x;
  if (idx >= tot) return;
  int c = idx & 255;
  int t = idx >> 8;
  int pos = t % (MGRID * MGRID);
  int b = t / (MGRID * MGRID);
  int my = pos >> 5, mx = pos & 31;
  float s = 0.f;
#pragma unroll
  for (int dy = 0; dy < 2; ++dy)
#pragma unroll
    for (int dx = 0; dx < 2; ++dx)
      s += A[((size_t)b * NTOK + (size_t)(my * 2 + dy) * HIMG + (mx * 2 + dx)) * 256 + c];
  P[((size_t)b * (MGRID * MGRID) + pos) * 256 + c] = 0.25f * s;
}

__global__ void upsample_add_kernel(const float* __restrict__ GL,
                                    const float* __restrict__ ATTN,
                                    float* __restrict__ SUM, int tot) {
  int idx = blockIdx.x * blockDim.x + threadIdx.x;
  if (idx >= tot) return;
  int c = idx & 255;
  int t = idx >> 8;
  int pos = t % NTOK;
  int b = t / NTOK;
  int y = pos >> 6, x = pos & 63;
  float sy = fminf(fmaxf((y + 0.5f) * 0.5f - 0.5f, 0.f), (float)(MGRID - 1));
  float sx = fminf(fmaxf((x + 0.5f) * 0.5f - 0.5f, 0.f), (float)(MGRID - 1));
  int y0 = (int)floorf(sy); int y1 = y0 + 1 < MGRID ? y0 + 1 : MGRID - 1; float fy = sy - (float)y0;
  int x0 = (int)floorf(sx); int x1 = x0 + 1 < MGRID ? x0 + 1 : MGRID - 1; float fx = sx - (float)x0;
  size_t base = (size_t)b * (MGRID * MGRID);
  float v00 = GL[(base + y0 * MGRID + x0) * 256 + c];
  float v01 = GL[(base + y0 * MGRID + x1) * 256 + c];
  float v10 = GL[(base + y1 * MGRID + x0) * 256 + c];
  float v11 = GL[(base + y1 * MGRID + x1) * 256 + c];
  float g = (1.f - fy) * ((1.f - fx) * v00 + fx * v01) + fy * ((1.f - fx) * v10 + fx * v11);
  size_t o = ((size_t)b * NTOK + pos) * 256 + c;
  SUM[o] = g + ATTN[o];
}

// ---------------------------------------------------------------------------
// Host driver (graph-capture safe: only kernel launches on `stream`).
// ---------------------------------------------------------------------------
extern "C" void kernel_launch(void* const* d_in, const int* in_sizes, int n_in,
                              void* d_out, int out_size, void* d_ws, size_t ws_size,
                              hipStream_t stream) {
  (void)in_sizes; (void)n_in; (void)out_size; (void)ws_size;
  const float* x     = (const float*)d_in[0];
  const float* Wq    = (const float*)d_in[1];
  const float* Wkv   = (const float*)d_in[2];
  const float* Wlkv  = (const float*)d_in[3];
  const float* Wproj = (const float*)d_in[4];
  const float* bproj = (const float*)d_in[5];
  float* out = (float*)d_out;

  char* wsb = (char*)d_ws;
  size_t off = 0;
  auto allocb = [&](size_t bytes) -> void* {
    void* p = (void*)(wsb + off);
    off = (off + bytes + 255) & ~(size_t)255;
    return p;
  };
  const int Mtok = BATCH * NTOK;           // 32768
  const int Mgl  = BATCH * MGRID * MGRID;  // 8192

  float* Q    = (float*)allocb(sizeof(float) * (size_t)Mtok * DIM);
  float* KV   = (float*)allocb(sizeof(float) * (size_t)Mtok * 2 * DIM);
  float* ATTN = (float*)allocb(sizeof(float) * (size_t)Mtok * DIM);
  float* QGT  = (float*)allocb(sizeof(float) * (size_t)Mgl * 32);
  float* Qg   = (float*)allocb(sizeof(float) * (size_t)Mgl * DIM);
  float* POOL = (float*)allocb(sizeof(float) * (size_t)Mgl * DIM);
  float* KV2  = (float*)allocb(sizeof(float) * (size_t)Mgl * 2 * DIM);
  float* GLA  = (float*)allocb(sizeof(float) * (size_t)Mgl * DIM);
  float* SUM  = (float*)allocb(sizeof(float) * (size_t)Mtok * DIM);
  _Float16* qh  = (_Float16*)allocb(sizeof(_Float16) * ((size_t)Mtok * 32 + 512));
  _Float16* kh  = (_Float16*)allocb(sizeof(_Float16) * ((size_t)Mtok * 32 + 512));
  _Float16* vth = (_Float16*)allocb(sizeof(_Float16) * ((size_t)Mtok * 32 + 512));
  _Float16* x16    = (_Float16*)allocb(sizeof(_Float16) * (size_t)Mtok * DIM);
  _Float16* P16    = (_Float16*)allocb(sizeof(_Float16) * (size_t)Mgl * DIM);
  _Float16* S16    = (_Float16*)allocb(sizeof(_Float16) * (size_t)Mtok * DIM);
  _Float16* Wqt    = (_Float16*)allocb(sizeof(_Float16) * (size_t)DIM * DIM);
  _Float16* Wkvt   = (_Float16*)allocb(sizeof(_Float16) * (size_t)DIM * 2 * DIM);
  _Float16* Wlkvt  = (_Float16*)allocb(sizeof(_Float16) * (size_t)DIM * 2 * DIM);
  _Float16* Wprojt = (_Float16*)allocb(sizeof(_Float16) * (size_t)DIM * DIM);

  const float scale = 0.17677669529663689f;  // 32^-0.5
  dim3 blk(256);

  // f16 operand preparation (A matrices row-major, weights transposed [N][K]).
  cvt_f16_kernel<<<dim3((Mtok * DIM + 255) / 256), blk, 0, stream>>>(x, x16, Mtok * DIM);
  transpose_f16_kernel<<<dim3((DIM * DIM + 255) / 256), blk, 0, stream>>>(Wq, Wqt, DIM, DIM, DIM * DIM);
  transpose_f16_kernel<<<dim3((DIM * 2 * DIM + 255) / 256), blk, 0, stream>>>(Wkv, Wkvt, DIM, 2 * DIM, DIM * 2 * DIM);
  transpose_f16_kernel<<<dim3((DIM * 2 * DIM + 255) / 256), blk, 0, stream>>>(Wlkv, Wlkvt, DIM, 2 * DIM, DIM * 2 * DIM);
  transpose_f16_kernel<<<dim3((DIM * DIM + 255) / 256), blk, 0, stream>>>(Wproj, Wprojt, DIM, DIM, DIM * DIM);

  // Q = x @ Wq ; KV = x @ Wkv   (TDM-staged WMMA GEMMs)
  wmma_gemm_f16_kernel<<<dim3(DIM / 64, Mtok / 128), blk, 0, stream>>>(x16, Wqt, nullptr, Q, Mtok, DIM, DIM);
  wmma_gemm_f16_kernel<<<dim3((2 * DIM) / 64, Mtok / 128), blk, 0, stream>>>(x16, Wkvt, nullptr, KV, Mtok, 2 * DIM, DIM);

  static const int REG[NHEADS] = {4, 4, 8, 8, 16, 16, 32, 32};
  for (int h = 0; h < NHEADS; ++h) {
    int p = REG[h], P2 = p * p, npx = HIMG / p, NP = npx * npx;
    int tot = Mtok * 32;
    pack_head_kernel<<<dim3((tot + 255) / 256), blk, 0, stream>>>(
        Q, KV, 2 * DIM, h * 32, h * 32, DIM + h * 32, qh, kh, vth, HIMG, p, tot);
    int tiles = BATCH * NP * (P2 / 16);  // == 2048 for every head
    attn_wmma_kernel<<<dim3((tiles + 7) / 8), blk, 0, stream>>>(
        qh, kh, vth, ATTN, h * 32, P2, NP, NTOK, scale, tiles);
    int totm = BATCH * P2 * 32;
    qg_mean_kernel<<<dim3((totm + 255) / 256), blk, 0, stream>>>(Q, h * 32, QGT, HIMG, p, totm);
    int totr = Mgl * 32;
    qg_resize_kernel<<<dim3((totr + 255) / 256), blk, 0, stream>>>(QGT, Qg, h * 32, p, totr);
  }

  // pooled = avgpool2x2(attn_fused); KV2 = pooled @ Wlkv
  int totp = Mgl * DIM;
  pool2x2_kernel<<<dim3((totp + 255) / 256), blk, 0, stream>>>(ATTN, POOL, totp);
  cvt_f16_kernel<<<dim3((totp + 255) / 256), blk, 0, stream>>>(POOL, P16, totp);
  wmma_gemm_f16_kernel<<<dim3((2 * DIM) / 64, Mgl / 128), blk, 0, stream>>>(P16, Wlkvt, nullptr, KV2, Mgl, 2 * DIM, DIM);

  // cross-attention at 32x32 (single patch covering the whole grid)
  for (int h = 0; h < NHEADS; ++h) {
    int tot = Mgl * 32;
    pack_head_kernel<<<dim3((tot + 255) / 256), blk, 0, stream>>>(
        Qg, KV2, 2 * DIM, h * 32, h * 32, DIM + h * 32, qh, kh, vth, MGRID, MGRID, tot);
    int tiles = BATCH * (MGRID * MGRID / 16);  // 512
    attn_wmma_kernel<<<dim3((tiles + 7) / 8), blk, 0, stream>>>(
        qh, kh, vth, GLA, h * 32, MGRID * MGRID, 1, MGRID * MGRID, scale, tiles);
  }

  // upsample + residual, final projection with bias
  int tots = Mtok * DIM;
  upsample_add_kernel<<<dim3((tots + 255) / 256), blk, 0, stream>>>(GLA, ATTN, SUM, tots);
  cvt_f16_kernel<<<dim3((tots + 255) / 256), blk, 0, stream>>>(SUM, S16, tots);
  wmma_gemm_f16_kernel<<<dim3(DIM / 64, Mtok / 128), blk, 0, stream>>>(S16, Wprojt, bproj, out, Mtok, DIM, DIM);
}
